// VectorQuantizer_13649406066973
// MI455X (gfx1250) — matrix operand order repaired and checked by reading, not compiled, not run on hardware
//
#include <hip/hip_runtime.h>

typedef __attribute__((ext_vector_type(16))) _Float16 v16h;
typedef __attribute__((ext_vector_type(8)))  float    v8f;
typedef __attribute__((ext_vector_type(4)))  float    v4f;
typedef __attribute__((ext_vector_type(4)))  int      v4i;

#define K_CODES 512
#define DEPTH   64
#define N_ROWS  65536
#define HSTRIDE 72                 // padded per-row stride (halfs): 144B, 16B aligned, conflict-free
#define ROWS_PER_BLOCK 128         // 8 waves x 16 rows
#define STAGE_ROWS 32              // codebook rows per async staging chunk (8KB)
#define OUT_Q_OFF 1
#define OUT_IDX_OFF (1 + N_ROWS * DEPTH)

#if defined(__has_builtin)
#  if __has_builtin(__builtin_amdgcn_global_load_async_to_lds_b128)
#    define HAVE_ASYNC_B128 1
#  endif
#  if __has_builtin(__builtin_amdgcn_s_wait_asynccnt)
#    define HAVE_WAIT_ASYNC 1
#  endif
#endif

__global__ void vq_init(float* ws) {
    if (threadIdx.x == 0) ws[0] = 0.0f;
}

__global__ __launch_bounds__(256)
void vq_main(const float* __restrict__ x, const float* __restrict__ cb,
             float* __restrict__ out, float* __restrict__ ws)
{
    extern __shared__ char smem[];
    _Float16* cb_hi = (_Float16*)smem;                          // 512*72 halfs
    _Float16* cb_lo = cb_hi + K_CODES * HSTRIDE;                // 512*72 halfs
    float*    cnorm = (float*)(cb_lo + K_CODES * HSTRIDE);      // 512 f32
    int*      s_idx = (int*)(cnorm + K_CODES);                  // 128 i32
    float*    s_red = (float*)(s_idx + ROWS_PER_BLOCK);         // 256 f32
    float*    stage = (float*)(s_red + 256);                    // 32*64 f32 = 8KB async staging

    const int t = threadIdx.x;

    // ---- Stage codebook: async global->LDS (ASYNCcnt path), then split to f16 hi/lo ----
#if defined(HAVE_ASYNC_B128)
    {
        typedef __attribute__((address_space(1))) v4i gv4i;   // global int4
        typedef __attribute__((address_space(3))) v4i lv4i;   // LDS int4
        const int sub = t & 7;          // 8 threads per codebook row
        const int r   = t >> 3;         // row within chunk (0..31)
        for (int chunk = 0; chunk < K_CODES / STAGE_ROWS; ++chunk) {
            char* gsrc = (char*)(cb + chunk * STAGE_ROWS * DEPTH);
            char* ldst = (char*)stage;
#pragma unroll
            for (int i = 0; i < 2; ++i) {           // 2 x 16B per thread = 8KB/block
                int off = (i * 256 + t) * 16;
                __builtin_amdgcn_global_load_async_to_lds_b128(
                    (gv4i*)(gsrc + off), (lv4i*)(ldst + off), 0, 0);
            }
#if defined(HAVE_WAIT_ASYNC)
            __builtin_amdgcn_s_wait_asynccnt(0);
#else
            asm volatile("s_wait_asynccnt 0x0" ::: "memory");
#endif
            __syncthreads();
            const int code = chunk * STAGE_ROWS + r;
            float nrm = 0.0f;
            const float* src = stage + r * DEPTH + sub * 8;
#pragma unroll
            for (int k = 0; k < 8; ++k) {
                float xv = src[k];
                nrm += xv * xv;
                _Float16 h = (_Float16)xv;
                cb_hi[code * HSTRIDE + sub * 8 + k] = h;
                cb_lo[code * HSTRIDE + sub * 8 + k] = (_Float16)(xv - (float)h);
            }
            nrm += __shfl_xor(nrm, 1, 32);
            nrm += __shfl_xor(nrm, 2, 32);
            nrm += __shfl_xor(nrm, 4, 32);
            if (sub == 0) cnorm[code] = nrm;
            __syncthreads();                        // staging buffer reused next chunk
        }
    }
#else
    for (int code = t; code < K_CODES; code += 256) {
        const float* src = cb + code * DEPTH;
        float nrm = 0.0f;
        for (int k = 0; k < DEPTH; k += 4) {
            v4f v = *(const v4f*)(src + k);
#pragma unroll
            for (int j = 0; j < 4; ++j) {
                float xv = v[j];
                nrm += xv * xv;
                _Float16 h = (_Float16)xv;
                cb_hi[code * HSTRIDE + k + j] = h;
                cb_lo[code * HSTRIDE + k + j] = (_Float16)(xv - (float)h);
            }
        }
        cnorm[code] = nrm;
    }
    __syncthreads();
#endif

    const int lane = t & 31;
    const int wave = t >> 5;
    const int g    = lane >> 4;    // half-wave group
    const int m    = lane & 15;    // row (A) / col (B,C,D) within tile
    const int rowTile = blockIdx.x * 8 + wave;
    const int r0 = rowTile * 16;

    // ---- Build A fragments (this code's contiguous k order, same as B: K = 32c + 16*g + 8*blk + j) ----
    v16h a_hi[2], a_lo[2];
    {
        const float* arow = x + (size_t)(r0 + m) * DEPTH + 16 * g;
#pragma unroll
        for (int c = 0; c < 2; ++c) {
#pragma unroll
            for (int blk = 0; blk < 2; ++blk) {
                v4f v0 = *(const v4f*)(arow + 32 * c + 8 * blk);
                v4f v1 = *(const v4f*)(arow + 32 * c + 8 * blk + 4);
#pragma unroll
                for (int j = 0; j < 4; ++j) {
                    float xv = v0[j];
                    _Float16 h = (_Float16)xv;
                    a_hi[c][blk * 8 + j] = h;
                    a_lo[c][blk * 8 + j] = (_Float16)(xv - (float)h);
                    xv = v1[j];
                    h = (_Float16)xv;
                    a_hi[c][blk * 8 + 4 + j] = h;
                    a_lo[c][blk * 8 + 4 + j] = (_Float16)(xv - (float)h);
                }
            }
        }
    }

    float bestVal[8];
    int   bestIdx[8];
#pragma unroll
    for (int v = 0; v < 8; ++v) { bestVal[v] = 3.0e38f; bestIdx[v] = 0; }

    // ---- Sweep codebook tiles: S = X * C^T via split-f16 WMMA, running argmin ----
#pragma unroll 2
    for (int ct = 0; ct < K_CODES / 16; ++ct) {
        const int code = ct * 16 + m;  // B fragment: this lane's codebook row
        const _Float16* rh = cb_hi + code * HSTRIDE + 16 * g;
        const _Float16* rl = cb_lo + code * HSTRIDE + 16 * g;
        v16h b_hi0 = *(const v16h*)(rh);
        v16h b_hi1 = *(const v16h*)(rh + 32);
        v16h b_lo0 = *(const v16h*)(rl);
        v16h b_lo1 = *(const v16h*)(rl + 32);

        v8f acc = {};
        acc = __builtin_amdgcn_wmma_f32_16x16x32_f16(false, a_hi[0], false, b_hi0, (short)0, acc, false, false);
        acc = __builtin_amdgcn_wmma_f32_16x16x32_f16(false, a_hi[1], false, b_hi1, (short)0, acc, false, false);
        acc = __builtin_amdgcn_wmma_f32_16x16x32_f16(false, a_lo[0], false, b_hi0, (short)0, acc, false, false);
        acc = __builtin_amdgcn_wmma_f32_16x16x32_f16(false, a_lo[1], false, b_hi1, (short)0, acc, false, false);
        acc = __builtin_amdgcn_wmma_f32_16x16x32_f16(false, a_hi[0], false, b_lo0, (short)0, acc, false, false);
        acc = __builtin_amdgcn_wmma_f32_16x16x32_f16(false, a_hi[1], false, b_lo1, (short)0, acc, false, false);

        const float cn = cnorm[code];
#pragma unroll
        for (int v = 0; v < 8; ++v) {
            float d = cn - 2.0f * acc[v];   // ||c||^2 - 2 x.c  (||x||^2 is row-constant)
            if (d < bestVal[v]) { bestVal[v] = d; bestIdx[v] = code; }
        }
    }

    // ---- Cross-lane argmin within each 16-lane half (tie-break: lowest index) ----
#pragma unroll
    for (int v = 0; v < 8; ++v) {
#pragma unroll
        for (int s = 1; s < 16; s <<= 1) {
            float ov = __shfl_xor(bestVal[v], s, 32);
            int   oi = __shfl_xor(bestIdx[v], s, 32);
            if (ov < bestVal[v] || (ov == bestVal[v] && oi < bestIdx[v])) {
                bestVal[v] = ov; bestIdx[v] = oi;
            }
        }
    }
    if (m == 0) {
#pragma unroll
        for (int v = 0; v < 8; ++v)
            s_idx[wave * 16 + g * 8 + v] = bestIdx[v];   // row = v + 8g per C-layout
    }
    __syncthreads();

    // ---- Gather quantized rows, write outputs, accumulate loss ----
    const size_t blockRow0 = (size_t)blockIdx.x * ROWS_PER_BLOCK;
    float lsum = 0.0f;
#pragma unroll 4
    for (int e = 0; e < 32; ++e) {
        int fi  = t + e * 256;         // 0..8191, coalesced
        int row = fi >> 6;
        int col = fi & 63;
        int code = s_idx[row];
        float q = cb[code * DEPTH + col];      // exact f32 codebook (L2-resident)
        size_t nidx = (blockRow0 + row) * DEPTH + col;
        float diff = q - x[nidx];
        lsum += diff * diff;
        out[OUT_Q_OFF + nidx] = q;             // quantized_st == quantized numerically
    }
    if (t < ROWS_PER_BLOCK)
        out[OUT_IDX_OFF + blockRow0 + t] = (float)s_idx[t];

    s_red[t] = lsum;
    __syncthreads();
    for (int s = 128; s > 0; s >>= 1) {
        if (t < s) s_red[t] += s_red[t + s];
        __syncthreads();
    }
    if (t == 0) atomicAdd(ws, s_red[0]);
}

__global__ void vq_fin(const float* ws, float* out) {
    // loss = q_latent + 0.25 * e_latent ; both equal mean((q-x)^2)
    if (threadIdx.x == 0) out[0] = 1.25f * ws[0] / 4194304.0f;
}

extern "C" void kernel_launch(void* const* d_in, const int* in_sizes, int n_in,
                              void* d_out, int out_size, void* d_ws, size_t ws_size,
                              hipStream_t stream) {
    const float* x  = (const float*)d_in[0];   // inputs  [64,32,32,64] f32
    const float* cb = (const float*)d_in[1];   // codebook [512,64]     f32
    float* out = (float*)d_out;
    float* ws  = (float*)d_ws;

    vq_init<<<1, 64, 0, stream>>>(ws);

    // NOTE: host pass cannot see device __has_builtin result; always reserve staging.
    size_t shmem = (size_t)(2 * K_CODES * HSTRIDE) * sizeof(_Float16)  // cb_hi + cb_lo
                 + (size_t)K_CODES * sizeof(float)                      // cnorm
                 + (size_t)ROWS_PER_BLOCK * sizeof(int)                 // s_idx
                 + 256 * sizeof(float)                                  // s_red
                 + (size_t)STAGE_ROWS * DEPTH * sizeof(float);          // async staging
    vq_main<<<N_ROWS / ROWS_PER_BLOCK, 256, shmem, stream>>>(x, cb, out, ws);

    vq_fin<<<1, 64, 0, stream>>>(ws, out);
}
